// GQA_28561532518475
// MI455X (gfx1250) — matrix operand order repaired and checked by reading, not compiled, not run on hardware
//
#include <hip/hip_runtime.h>

// ---------------------------------------------------------------------------
// GQA forward for MI455X (gfx1250): bf16 WMMA everywhere, fp32 softmax.
// Shapes: B=4, P=2048, C=768, H=12, HK=6, HD=64, GS=2.
// Round 4: attention K/V staged into LDS by the Tensor Data Mover
// (tensor_load_to_lds + s_wait_tensorcnt), double-buffered, shared by the
// 4 waves of a workgroup (same (b,h), 4 consecutive 32-row q tiles).
// ---------------------------------------------------------------------------

typedef __attribute__((ext_vector_type(16))) __bf16 v16bf;
typedef __attribute__((ext_vector_type(8)))  __bf16 v8bf;
typedef __attribute__((ext_vector_type(8)))  float  v8f;
typedef __attribute__((ext_vector_type(4)))  unsigned int v4u;
typedef __attribute__((ext_vector_type(4)))  int  v4i;
typedef __attribute__((ext_vector_type(8)))  int  v8i;

#define WMMA_BF16(A, Bm, Cm) \
  __builtin_amdgcn_wmma_f32_16x16x32_bf16(false, (A), false, (Bm), (short)0, (Cm), false, false)

// SCALE * log2(e) folded into Q so softmax can use exp2 (v_exp_f32).
__device__ __host__ constexpr float kScaleLog2e = 0.125f * 1.44269504088896340736f;

// ---------------------------------------------------------------------------
// Fragment loaders per CDNA5 ISA 7.12.2 layouts (wave32).
// ---------------------------------------------------------------------------
__device__ inline v16bf load_a_bf16(const __bf16* A, int lda, int lane) {
  int m  = lane & 15;
  int kb = (lane >> 4) << 3;          // 0 or 8
  const __bf16* p = A + (size_t)m * lda;
  v8bf lo = *(const v8bf*)(p + kb);
  v8bf hi = *(const v8bf*)(p + 16 + kb);
  return __builtin_shufflevector(lo, hi, 0,1,2,3,4,5,6,7,8,9,10,11,12,13,14,15);
}

// B 32x16 bf16 (KxN), stored K-major per column: Bt[n][k], ld = K stride.
__device__ inline v16bf load_b_bf16(const __bf16* Bt, int ldb, int lane) {
  int n  = lane & 15;
  int kb = (lane >> 4) << 3;          // 0 or 8
  const __bf16* p = Bt + (size_t)n * ldb + kb;
  v8bf lo = *(const v8bf*)(p);
  v8bf hi = *(const v8bf*)(p + 16);
  return __builtin_shufflevector(lo, hi, 0,1,2,3,4,5,6,7,8,9,10,11,12,13,14,15);
}

// ---------------------------------------------------------------------------
// Tensor Data Mover: issue one 2D tile load (global -> LDS).
// D# group0: [1:0]=count=1, [63:32]=lds_addr, [120:64]=global_addr,
// [127:126]=type=2. Groups 2/3 zero => 2D tensor (tile_dim2=0 unused).
// ---------------------------------------------------------------------------
__device__ inline void tdm_load_2d(unsigned lds_addr, unsigned long long ga, v8i g1) {
  v4u g0;
  g0[0] = 1u;                                        // count=1, user descriptor
  g0[1] = lds_addr;                                  // LDS byte address
  g0[2] = (unsigned)(ga & 0xffffffffull);            // global addr [31:0]
  g0[3] = (unsigned)((ga >> 32) & 0x1ffffffull)      // global addr [56:32]
        | (2u << 30);                                // type = 2 ("image")
  v4i z4 = {0, 0, 0, 0};
#if defined(__clang_major__) && (__clang_major__ >= 23)
  v8i z8 = {0, 0, 0, 0, 0, 0, 0, 0};
  __builtin_amdgcn_tensor_load_to_lds(g0, g1, z4, z4, z8, 0);
#else
  __builtin_amdgcn_tensor_load_to_lds(g0, g1, z4, z4, 0);
#endif
}

// ---------------------------------------------------------------------------
// Prep kernels: fp32 -> bf16 conversions with permutation folding.
// ---------------------------------------------------------------------------
__global__ void conv_x_kernel(const float* __restrict__ x, __bf16* __restrict__ xb, int n) {
  int i = blockIdx.x * blockDim.x + threadIdx.x;
  if (i < n) xb[i] = (__bf16)x[i];
}

__global__ void prep_wq_kernel(const float* __restrict__ Wq, const int* __restrict__ perm,
                               __bf16* __restrict__ Wqt) {
  int idx = blockIdx.x * blockDim.x + threadIdx.x;
  if (idx >= 768 * 768) return;
  int n = idx / 768, k = idx % 768;
  int src_col = perm[n >> 6] * 64 + (n & 63);
  Wqt[idx] = (__bf16)Wq[(size_t)k * 768 + src_col];
}

__global__ void prep_wkv_kernel(const float* __restrict__ W, const int* __restrict__ perm,
                                __bf16* __restrict__ Wt) {
  int idx = blockIdx.x * blockDim.x + threadIdx.x;
  if (idx >= 384 * 768) return;
  int n = idx / 768, k = idx % 768;
  int hb = k >> 6;
  int inv = 0;
  #pragma unroll
  for (int j = 0; j < 12; ++j) if (perm[j] == hb) inv = j;
  int src_row = inv * 64 + (k & 63);
  Wt[idx] = (__bf16)W[(size_t)src_row * 384 + n];
}

__global__ void prep_wp_kernel(const float* __restrict__ Wp, __bf16* __restrict__ Wpt) {
  int idx = blockIdx.x * blockDim.x + threadIdx.x;
  if (idx >= 768 * 768) return;
  int n = idx / 768, k = idx % 768;
  Wpt[idx] = (__bf16)Wp[(size_t)k * 768 + n];
}

// ---------------------------------------------------------------------------
// Fused QKV GEMM: [8192 x 768] x [768 x 1536].
// One wave computes a 32(M) x 64(N) tile; K-loop unrolled x2 with ping-pong
// fragment buffers (no rotation copies).
// ---------------------------------------------------------------------------
__global__ __launch_bounds__(128) void qkv_kernel(
    const __bf16* __restrict__ xb,   // [8192][768]
    const __bf16* __restrict__ Wqt,  // [768][768]
    const __bf16* __restrict__ Wkt,  // [384][768]
    const __bf16* __restrict__ Wvt,  // [384][768]
    __bf16* __restrict__ Qb,         // [48][2048][64]
    __bf16* __restrict__ Kb,         // [24][2048][64]
    __bf16* __restrict__ Vt)         // [24][64][2048]
{
  int lane = threadIdx.x & 31;
  int wid  = blockIdx.x * 4 + (threadIdx.x >> 5);
  int nt = wid % 24;                 // 64-wide N tile
  int mt = wid / 24;                 // 32-wide M tile (0..255)
  int m0 = mt * 32;
  int n0 = nt * 64;

  const __bf16* Bt; int bn0;
  if (n0 < 768)        { Bt = Wqt; bn0 = n0; }
  else if (n0 < 1152)  { Bt = Wkt; bn0 = n0 - 768; }
  else                 { Bt = Wvt; bn0 = n0 - 1152; }

  const __bf16* Arow0 = xb + (size_t)m0 * 768;
  const __bf16* Arow1 = xb + (size_t)(m0 + 16) * 768;

  v8f acc[2][4] = {};

  v16bf a0A, a1A, bA[4];             // buffer A
  v16bf a0B, a1B, bB[4];             // buffer B

  auto load_frags = [&](int k, v16bf& fa0, v16bf& fa1, v16bf* fb) {
    fa0 = load_a_bf16(Arow0 + k, 768, lane);
    fa1 = load_a_bf16(Arow1 + k, 768, lane);
    #pragma unroll
    for (int t = 0; t < 4; ++t)
      fb[t] = load_b_bf16(Bt + (size_t)(bn0 + t * 16) * 768 + k, 768, lane);
  };
  auto compute = [&](const v16bf& fa0, const v16bf& fa1, const v16bf* fb) {
    #pragma unroll
    for (int t = 0; t < 4; ++t) {
      acc[0][t] = WMMA_BF16(fa0, fb[t], acc[0][t]);
      acc[1][t] = WMMA_BF16(fa1, fb[t], acc[1][t]);
    }
  };

  load_frags(0, a0A, a1A, bA);
  for (int k0 = 0; k0 < 768; k0 += 64) {
    load_frags(k0 + 32, a0B, a1B, bB);
    compute(a0A, a1A, bA);
    load_frags((k0 + 64) % 768, a0A, a1A, bA);   // wraps to 0 on last iter (unused)
    compute(a0B, a1B, bB);
  }

  int half = lane >> 4, col = lane & 15;
  int bidx = m0 >> 11;               // batch (tiles never cross batch: 2048%32==0)

  #pragma unroll
  for (int su = 0; su < 2; ++su) {
    int prow = (m0 & 2047) + su * 16 + half * 8;
    if (n0 < 768) {                  // Q, pre-scaled by SCALE*log2e
      #pragma unroll
      for (int t = 0; t < 4; ++t) {
        int n = n0 + t * 16 + col, hd = n >> 6, d = n & 63;
        #pragma unroll
        for (int r = 0; r < 8; ++r)
          Qb[(((size_t)(bidx * 12 + hd)) * 2048 + prow + r) * 64 + d] =
              (__bf16)(acc[su][t][r] * kScaleLog2e);
      }
    } else if (n0 < 1152) {          // K -> [b,kh,p,d]
      #pragma unroll
      for (int t = 0; t < 4; ++t) {
        int n = n0 - 768 + t * 16 + col, kh = n >> 6, d = n & 63;
        #pragma unroll
        for (int r = 0; r < 8; ++r)
          Kb[(((size_t)(bidx * 6 + kh)) * 2048 + prow + r) * 64 + d] = (__bf16)acc[su][t][r];
      }
    } else {                         // V -> transposed [b,kh,d,p]
      #pragma unroll
      for (int t = 0; t < 4; ++t) {
        int n = n0 - 1152 + t * 16 + col, kh = n >> 6, d = n & 63;
        v8bf pk;
        #pragma unroll
        for (int r = 0; r < 8; ++r) pk[r] = (__bf16)acc[su][t][r];
        *(v8bf*)(Vt + (((size_t)(bidx * 6 + kh)) * 64 + d) * 2048 + prow) = pk;
      }
    }
  }
}

// ---------------------------------------------------------------------------
// Flash attention with TDM-staged K/V.
// Block = 4 waves sharing one (b,h); wave w owns q rows [qt*32, qt*32+32).
// Wave 0 drives the Tensor Data Mover: per 32-key chunk it DMAs
//   K tile  [32 keys][64 d]  (4 KB) and  V^T tile [64 d][32 keys] (4 KB)
// into double-buffered LDS; s_wait_tensorcnt + barriers sequence the buffers.
// All waves read WMMA B-fragments straight from LDS (ds_load_b128).
// ---------------------------------------------------------------------------
struct AttnShared {
  __bf16 kbuf[2][32 * 64];   // [key][d]
  __bf16 vbuf[2][64 * 32];   // [d][key]
  __bf16 probs[4][32 * 32];  // per-wave P transpose staging
};

__global__ __launch_bounds__(128) void attn_kernel(
    const __bf16* __restrict__ Qb,   // [48][2048][64], pre-scaled
    const __bf16* __restrict__ Kb,   // [24][2048][64]
    const __bf16* __restrict__ Vt,   // [24][64][2048]
    const int* __restrict__ perm,
    __bf16* __restrict__ Ob)         // [4][2048][768] bf16
{
  __shared__ __align__(16) AttnShared sh;
  int lane = threadIdx.x & 31;
  int w    = threadIdx.x >> 5;
  int bh = blockIdx.x >> 4;          // 0..47 : (b, permuted head)
  int qg = blockIdx.x & 15;          // q-group of 128 rows
  int qt = qg * 4 + w;               // this wave's 32-row q tile (0..63)
  int b = bh / 12, hd = bh % 12;
  int kh = hd >> 1;                  // GS = 2
  int half = lane >> 4, col = lane & 15;

  const __bf16* Qtile = Qb + (((size_t)bh) * 2048 + qt * 32) * 64;
  v16bf qa[2][2];
  #pragma unroll
  for (int su = 0; su < 2; ++su) {
    qa[su][0] = load_a_bf16(Qtile + (size_t)su * 16 * 64 +  0, 64, lane);
    qa[su][1] = load_a_bf16(Qtile + (size_t)su * 16 * 64 + 32, 64, lane);
  }

  const __bf16* Kbase = Kb + ((size_t)(b * 6 + kh)) * 2048 * 64;
  const __bf16* Vbase = Vt + ((size_t)(b * 6 + kh)) * 64 * 2048;
  __bf16* myl = sh.probs[w];

  // TDM group-1 descriptors (data_size=1 -> 2 bytes/elem).
  // K: tensor 2048x64 row-major, stride 64; tile 64(d) x 32(keys).
  v8i g1K = { (int)(1u << 16), (int)(64u << 16), (int)(2048u << 16),
              (int)(64u << 16), 32, 64, 0, 0 };
  // V: tensor 64x2048 row-major, stride 2048; tile 32(keys) x 64(d rows).
  v8i g1V = { (int)(1u << 16), (int)(2048u << 16), (int)(64u << 16),
              (int)(32u << 16), 64, 2048, 0, 0 };

  // LDS byte addresses: flat shared pointers carry the LDS offset in [31:0].
  unsigned kldsa[2] = { (unsigned)(size_t)&sh.kbuf[0][0], (unsigned)(size_t)&sh.kbuf[1][0] };
  unsigned vldsa[2] = { (unsigned)(size_t)&sh.vbuf[0][0], (unsigned)(size_t)&sh.vbuf[1][0] };

  auto issue_chunk = [&](int kc, int buf) {
    tdm_load_2d(kldsa[buf], (unsigned long long)(size_t)(Kbase + (size_t)kc * 64), g1K);
    tdm_load_2d(vldsa[buf], (unsigned long long)(size_t)(Vbase + kc), g1V);
  };

  float mrun[2][8], lrun[2][8];
  #pragma unroll
  for (int su = 0; su < 2; ++su)
    #pragma unroll
    for (int r = 0; r < 8; ++r) { mrun[su][r] = -1e30f; lrun[su][r] = 0.0f; }
  v8f o[2][4] = {};

  // Consume one staged 32-key chunk from LDS buffer `buf`.
  auto process = [&](int buf) {
    const __bf16* ktile = sh.kbuf[buf];
    const __bf16* vtile = sh.vbuf[buf];

    // K B-fragments from LDS: subtile t keys, d-half h.
    v16bf kbf[2][2];
    #pragma unroll
    for (int t = 0; t < 2; ++t)
      #pragma unroll
      for (int h = 0; h < 2; ++h)
        kbf[t][h] = load_b_bf16(ktile + (size_t)(t * 16) * 64 + h * 32, 64, lane);

    // ---- scores: 2 q-subtiles x 2 key-subtiles, K-dim 64 = 2 WMMAs each
    v8f s[2][2] = {};
    #pragma unroll
    for (int su = 0; su < 2; ++su)
      #pragma unroll
      for (int t = 0; t < 2; ++t) {
        s[su][t] = WMMA_BF16(qa[su][0], kbf[t][0], s[su][t]);
        s[su][t] = WMMA_BF16(qa[su][1], kbf[t][1], s[su][t]);
      }

    // ---- V B-fragments from LDS (issued before softmax VALU block)
    v16bf vb[4];
    #pragma unroll
    for (int t = 0; t < 4; ++t)
      vb[t] = load_b_bf16(vtile + (size_t)(t * 16) * 32, 32, lane);

    // ---- online softmax (fp32)
    #pragma unroll
    for (int su = 0; su < 2; ++su) {
      #pragma unroll
      for (int r = 0; r < 8; ++r) {
        float mx = fmaxf(s[su][0][r], s[su][1][r]);
        #pragma unroll
        for (int off = 1; off < 16; off <<= 1) mx = fmaxf(mx, __shfl_xor(mx, off, 32));
        float nm    = fmaxf(mrun[su][r], mx);
        float alpha = exp2f(mrun[su][r] - nm);
        float p0    = exp2f(s[su][0][r] - nm);
        float p1    = exp2f(s[su][1][r] - nm);
        float rs = p0 + p1;
        #pragma unroll
        for (int off = 1; off < 16; off <<= 1) rs += __shfl_xor(rs, off, 32);
        lrun[su][r] = lrun[su][r] * alpha + rs;
        mrun[su][r] = nm;
        #pragma unroll
        for (int t = 0; t < 4; ++t) o[su][t][r] *= alpha;
        int row = su * 16 + r + half * 8;
        myl[row * 32 + col]      = (__bf16)p0;
        myl[row * 32 + 16 + col] = (__bf16)p1;
      }
    }

    // ---- P (32x32) back as A-fragments, multiply by V^T B-fragments (N=64)
    #pragma unroll
    for (int su = 0; su < 2; ++su) {
      v16bf pa = load_a_bf16(myl + su * 16 * 32, 32, lane);
      #pragma unroll
      for (int t = 0; t < 4; ++t)
        o[su][t] = WMMA_BF16(pa, vb[t], o[su][t]);
    }
  };

  // ---- TDM-pipelined main loop over 64 chunks of 32 keys
  if (w == 0) issue_chunk(0, 0);
  for (int i = 0; i < 64; ++i) {
    if (w == 0) {
      if (i < 63) {
        issue_chunk((i + 1) * 32, (i + 1) & 1);
        __builtin_amdgcn_s_wait_tensorcnt(2);  // chunk i landed (in-order)
      } else {
        __builtin_amdgcn_s_wait_tensorcnt(0);
      }
    }
    __syncthreads();                 // chunk i visible to all waves
    process(i & 1);
    __syncthreads();                 // everyone done before buffer reuse
  }

  // ---- epilogue: normalize, write to permuted head slot
  int head_out = perm[hd];
  #pragma unroll
  for (int su = 0; su < 2; ++su)
    #pragma unroll
    for (int t = 0; t < 4; ++t) {
      int d = t * 16 + col;
      #pragma unroll
      for (int r = 0; r < 8; ++r) {
        int p = qt * 32 + su * 16 + r + half * 8;
        float val = o[su][t][r] / lrun[su][r];
        Ob[(((size_t)b * 2048) + p) * 768 + head_out * 64 + d] = (__bf16)val;
      }
    }
}

// ---------------------------------------------------------------------------
// Output projection: [8192 x 768] x [768 x 768] + bias -> fp32 out.
// 32x64 tile per wave; ping-pong K-loop.
// ---------------------------------------------------------------------------
__global__ __launch_bounds__(128) void proj_kernel(
    const __bf16* __restrict__ Ob,   // [8192][768]
    const __bf16* __restrict__ Wpt,  // [768][768] (transposed)
    const float* __restrict__ bp,
    float* __restrict__ out)
{
  int lane = threadIdx.x & 31;
  int wid  = blockIdx.x * 4 + (threadIdx.x >> 5);
  int nt = wid % 12;
  int mt = wid / 12;                 // 0..255
  int m0 = mt * 32, n0 = nt * 64;

  const __bf16* Arow0 = Ob + (size_t)m0 * 768;
  const __bf16* Arow1 = Ob + (size_t)(m0 + 16) * 768;

  v8f acc[2][4] = {};
  v16bf a0A, a1A, bA[4];
  v16bf a0B, a1B, bB[4];

  auto load_frags = [&](int k, v16bf& fa0, v16bf& fa1, v16bf* fb) {
    fa0 = load_a_bf16(Arow0 + k, 768, lane);
    fa1 = load_a_bf16(Arow1 + k, 768, lane);
    #pragma unroll
    for (int t = 0; t < 4; ++t)
      fb[t] = load_b_bf16(Wpt + (size_t)(n0 + t * 16) * 768 + k, 768, lane);
  };
  auto compute = [&](const v16bf& fa0, const v16bf& fa1, const v16bf* fb) {
    #pragma unroll
    for (int t = 0; t < 4; ++t) {
      acc[0][t] = WMMA_BF16(fa0, fb[t], acc[0][t]);
      acc[1][t] = WMMA_BF16(fa1, fb[t], acc[1][t]);
    }
  };

  load_frags(0, a0A, a1A, bA);
  for (int k0 = 0; k0 < 768; k0 += 64) {
    load_frags(k0 + 32, a0B, a1B, bB);
    compute(a0A, a1A, bA);
    load_frags((k0 + 64) % 768, a0A, a1A, bA);
    compute(a0B, a1B, bB);
  }

  int half = lane >> 4, col = lane & 15;
  #pragma unroll
  for (int su = 0; su < 2; ++su)
    #pragma unroll
    for (int t = 0; t < 4; ++t) {
      int n = n0 + t * 16 + col;
      float bias = bp[n];
      #pragma unroll
      for (int r = 0; r < 8; ++r) {
        int m = m0 + su * 16 + r + half * 8;
        out[(size_t)m * 768 + n] = acc[su][t][r] + bias;
      }
    }
}

// ---------------------------------------------------------------------------
extern "C" void kernel_launch(void* const* d_in, const int* in_sizes, int n_in,
                              void* d_out, int out_size, void* d_ws, size_t ws_size,
                              hipStream_t stream)
{
  const float* x   = (const float*)d_in[0];
  const float* Wq  = (const float*)d_in[1];
  const float* Wk  = (const float*)d_in[2];
  const float* Wv  = (const float*)d_in[3];
  const float* Wp  = (const float*)d_in[4];
  const float* bp  = (const float*)d_in[5];
  const int*  perm = (const int*)d_in[6];
  float* out = (float*)d_out;
  (void)in_sizes; (void)n_in; (void)out_size; (void)ws_size;

  char* ws = (char*)d_ws;
  size_t off = 0;
  auto take = [&](size_t bytes) -> char* {
    char* p = ws + off;
    off = (off + bytes + 255) & ~(size_t)255;
    return p;
  };
  __bf16* xb  = (__bf16*)take(8192ull * 768 * 2);
  __bf16* Wqt = (__bf16*)take(768ull * 768 * 2);
  __bf16* Wkt = (__bf16*)take(384ull * 768 * 2);
  __bf16* Wvt = (__bf16*)take(384ull * 768 * 2);
  __bf16* Wpt = (__bf16*)take(768ull * 768 * 2);
  __bf16* Qb  = (__bf16*)take(48ull * 2048 * 64 * 2);
  __bf16* Kb  = (__bf16*)take(24ull * 2048 * 64 * 2);
  __bf16* Vt  = (__bf16*)take(24ull * 2048 * 64 * 2);
  __bf16* Ob  = (__bf16*)take(8192ull * 768 * 2);

  conv_x_kernel <<<(8192 * 768 + 255) / 256, 256, 0, stream>>>(x, xb, 8192 * 768);
  prep_wq_kernel<<<(768 * 768 + 255) / 256, 256, 0, stream>>>(Wq, perm, Wqt);
  prep_wkv_kernel<<<(384 * 768 + 255) / 256, 256, 0, stream>>>(Wk, perm, Wkt);
  prep_wkv_kernel<<<(384 * 768 + 255) / 256, 256, 0, stream>>>(Wv, perm, Wvt);
  prep_wp_kernel<<<(768 * 768 + 255) / 256, 256, 0, stream>>>(Wp, Wpt);

  // 256 M-tiles x 24 N-tiles = 6144 waves, 4 waves/block
  qkv_kernel<<<1536, 128, 0, stream>>>(xb, Wqt, Wkt, Wvt, Qb, Kb, Vt);
  // 48 (b,h) x 16 q-groups = 768 blocks; 4 waves/block share TDM-staged K/V
  attn_kernel<<<768, 128, 0, stream>>>(Qb, Kb, Vt, perm, Ob);
  // 256 M-tiles x 12 N-tiles = 3072 waves
  proj_kernel<<<768, 128, 0, stream>>>(Ob, Wpt, bp, out);
}
